// RoPEMultiheadAttention_6820408066544
// MI455X (gfx1250) — matrix operand order repaired and checked by reading, not compiled, not run on hardware
//
#include <hip/hip_runtime.h>
#include <hip/hip_bf16.h>

// ---------------------------------------------------------------------------
// RoPE multi-head attention on v_wmma_f32_16x16x32_bf16 (gfx1250, wave32).
// B=2, N=2048, H=12, D=64, C=768. fp32 in/out, bf16 matrix operands,
// fp32 accumulation + fp32 online softmax + fp32 RoPE.
//  - GEMMs: 16x64 output per wave (A-fragment reused across 4 WMMAs)
//  - Attention: 32 queries per wave (K/V fragments reused across 2 S / 2 PV)
// ---------------------------------------------------------------------------

typedef __attribute__((ext_vector_type(16))) __bf16        v16bf;
typedef __attribute__((ext_vector_type(8)))  float         v8f;
typedef __attribute__((ext_vector_type(4)))  unsigned int  v4u;

union Frag {
    v16bf v;
    v4u   q[2];   // two 16-byte chunks
};

#define BATCH   2
#define SEQ     2048
#define HEADS   12
#define HDIM    64
#define EMB     768          // HEADS*HDIM
#define QKVC    2304         // 3*EMB
#define ROWS    4096         // BATCH*SEQ

// ---------------------------------------------------------------------------
// fp32 -> bf16 elementwise
__global__ void f32_to_bf16_kernel(const float* __restrict__ in,
                                   __bf16* __restrict__ out, int n) {
    int i = blockIdx.x * blockDim.x + threadIdx.x;
    if (i < n) out[i] = (__bf16)in[i];
}

// fp32 (R x Cc) -> bf16 transposed (Cc x R):  out[c*R + r] = in[r*Cc + c]
__global__ void transpose_to_bf16_kernel(const float* __restrict__ in,
                                         __bf16* __restrict__ out,
                                         int R, int Cc) {
    int i = blockIdx.x * blockDim.x + threadIdx.x;
    if (i >= R * Cc) return;
    int r = i / Cc, c = i % Cc;
    out[c * R + r] = (__bf16)in[r * Cc + c];
}

// ---------------------------------------------------------------------------
// qkv = x @ Wqkv, +RoPE on q,k.  One wave -> one 16x64 output tile (4 accs).
//   xb : ROWS x EMB (bf16, row major)
//   wT : QKVC x EMB (bf16, Wqkv transposed -> contiguous K per output column)
// Outputs: Qr,Kr as (B,H,N,D) bf16 (RoPE applied); Vt as (B,H,D,N) bf16.
__global__ __launch_bounds__(32)
void qkv_rope_kernel(const __bf16* __restrict__ xb,
                     const __bf16* __restrict__ wT,
                     const float*  __restrict__ cosp,   // SEQ*HDIM
                     const float*  __restrict__ sinp,   // SEQ*HDIM
                     __bf16* __restrict__ Qr,
                     __bf16* __restrict__ Kr,
                     __bf16* __restrict__ Vt) {
    const int lane = threadIdx.x;
    const int hi   = lane >> 4;        // 0 / 1
    const int nloc = lane & 15;
    const int colBase = blockIdx.x * 64;   // [0, 2304)
    const int rowBase = blockIdx.y * 16;   // [0, 4096)

    const __bf16* aptr  = xb + (size_t)(rowBase + nloc) * EMB;   // A row = lane%16
    const __bf16* bptr0 = wT + (size_t)(colBase + nloc) * EMB;   // B col base
    const int ak = hi * 16;   // A K-offset:  {0..15} vs {16..31}
    const int bk = hi * 16;   // B K-offset:  {0..15} vs {16..31}

    v8f acc[4] = {v8f{}, v8f{}, v8f{}, v8f{}};
    for (int kk = 0; kk < EMB; kk += 32) {
        Frag a;
        a.q[0] = *(const v4u*)(aptr + kk + ak);
        a.q[1] = *(const v4u*)(aptr + kk + ak + 8);
        __builtin_prefetch(aptr + kk + 256, 0, 0);      // global_prefetch_b8
        #pragma unroll
        for (int g = 0; g < 4; ++g) {
            const __bf16* bp = bptr0 + (size_t)g * 16 * EMB + kk + bk;
            Frag b;
            b.q[0] = *(const v4u*)(bp);
            b.q[1] = *(const v4u*)(bp + 8);
            acc[g] = __builtin_amdgcn_wmma_f32_16x16x32_bf16(
                false, a.v, false, b.v, (short)0, acc[g], false, false);
        }
    }

    // Epilogue: C layout -> row M = r + 8*hi, col = colBase + g*16 + nloc
    #pragma unroll
    for (int g = 0; g < 4; ++g) {
        #pragma unroll
        for (int r = 0; r < 8; ++r) {
            float val = acc[g][r];
            float prt = __shfl_xor(val, 1);      // RoPE partner d^1 (lane^1)
            int row = rowBase + r + 8 * hi;      // [0,4096)
            int b   = row >> 11;                 // /2048
            int n   = row & 2047;
            int col = colBase + g * 16 + nloc;   // [0,2304); section uniform/wave
            if (col < 2 * EMB) {                 // q or k -> apply RoPE
                int d  = col & 63;
                float cs = cosp[(n << 6) + d];
                float sn = sinp[(n << 6) + d];
                float rot = (d & 1) ? prt : -prt;   // rotate_half (interleaved)
                float o = val * cs + rot * sn;
                int c = (col < EMB) ? col : (col - EMB);
                int h = c >> 6;
                __bf16* dst = (col < EMB) ? Qr : Kr;
                dst[(((size_t)(b * HEADS + h) << 11) + n) * HDIM + d] = (__bf16)o;
            } else {                             // v -> store transposed (B,H,D,N)
                int c = col - 2 * EMB;
                int h = c >> 6, d = c & 63;
                Vt[((size_t)(b * HEADS + h) * HDIM + d) * SEQ + n] = (__bf16)val;
            }
        }
    }
}

// ---------------------------------------------------------------------------
// Flash attention: one wave handles 32 queries (two 16-row tiles) of one
// (b,h), streaming all 2048 keys in blocks of 32.  Per key block:
// 16 WMMAs (8 for S, 8 for P@V) against 16 b128 loads (K and V shared).
__global__ __launch_bounds__(32)
void attn_kernel(const __bf16* __restrict__ Qr,
                 const __bf16* __restrict__ Kr,
                 const __bf16* __restrict__ Vt,
                 __bf16* __restrict__ AO) {          // ROWS x EMB bf16
    __shared__ alignas(16) __bf16 Plds[2][16][32];   // P staging (C -> A layout)

    const int lane = threadIdx.x;
    const int hi   = lane >> 4;
    const int nloc = lane & 15;
    const int q0   = blockIdx.x * 32;    // first query row of this wave
    const int bh   = blockIdx.y;         // [0,24)
    const int b    = bh / HEADS, h = bh % HEADS;
    const size_t kqbase = (size_t)bh * SEQ * HDIM;   // (B,H,N,D)
    const size_t vbase  = (size_t)bh * HDIM * SEQ;   // (B,H,D,N)

    // Q fragments: two query tiles x two K-chunks covering D=64
    Frag qf[2][2];
    #pragma unroll
    for (int t = 0; t < 2; ++t) {
        const __bf16* qp = Qr + kqbase + (size_t)(q0 + t * 16 + nloc) * HDIM + hi * 16;
        qf[t][0].q[0] = *(const v4u*)(qp);
        qf[t][0].q[1] = *(const v4u*)(qp + 8);
        qf[t][1].q[0] = *(const v4u*)(qp + 32);
        qf[t][1].q[1] = *(const v4u*)(qp + 40);
    }

    float m[2][8], l[2][8];
    v8f accO[2][4];
    #pragma unroll
    for (int t = 0; t < 2; ++t)
        #pragma unroll
        for (int r = 0; r < 8; ++r) { m[t][r] = -1e30f; l[t][r] = 0.0f; }
    #pragma unroll
    for (int t = 0; t < 2; ++t)
        #pragma unroll
        for (int g = 0; g < 4; ++g) accO[t][g] = v8f{};

    for (int kb = 0; kb < SEQ; kb += 32) {
        // ---- K fragments for 32 keys (shared by both query tiles) ----
        Frag kf[2][2];                       // [16-key half][d-chunk]
        #pragma unroll
        for (int c = 0; c < 2; ++c) {
            const __bf16* kp = Kr + kqbase +
                               (size_t)(kb + c * 16 + nloc) * HDIM + hi * 16;
            kf[c][0].q[0] = *(const v4u*)(kp);
            kf[c][0].q[1] = *(const v4u*)(kp + 8);
            kf[c][1].q[0] = *(const v4u*)(kp + 32);
            kf[c][1].q[1] = *(const v4u*)(kp + 40);
        }

        __syncthreads();   // previous iteration's P-fragment reads are done
        #pragma unroll
        for (int t = 0; t < 2; ++t) {
            // ---- S = Q @ K^T, 16x32 tile as two column halves ----
            v8f S[2] = {v8f{}, v8f{}};
            #pragma unroll
            for (int c = 0; c < 2; ++c) {
                S[c] = __builtin_amdgcn_wmma_f32_16x16x32_bf16(
                    false, qf[t][0].v, false, kf[c][0].v, (short)0, S[c], false, false);
                S[c] = __builtin_amdgcn_wmma_f32_16x16x32_bf16(
                    false, qf[t][1].v, false, kf[c][1].v, (short)0, S[c], false, false);
            }
            // ---- fp32 online softmax (rows live across each 16-lane half) ----
            #pragma unroll
            for (int r = 0; r < 8; ++r) {
                float s0 = S[0][r] * 0.125f;     // 1/sqrt(64)
                float s1 = S[1][r] * 0.125f;
                float mx = fmaxf(s0, s1);
                #pragma unroll
                for (int off = 1; off < 16; off <<= 1)
                    mx = fmaxf(mx, __shfl_xor(mx, off));
                float mn    = fmaxf(m[t][r], mx);
                float alpha = __expf(m[t][r] - mn);
                float p0 = __expf(s0 - mn);
                float p1 = __expf(s1 - mn);
                float sum = p0 + p1;
                #pragma unroll
                for (int off = 1; off < 16; off <<= 1)
                    sum += __shfl_xor(sum, off);
                l[t][r] = l[t][r] * alpha + sum;
                m[t][r] = mn;
                #pragma unroll
                for (int g = 0; g < 4; ++g) accO[t][g][r] *= alpha;
                int M = r + 8 * hi;              // P: C layout -> LDS
                Plds[t][M][nloc]      = (__bf16)p0;
                Plds[t][M][16 + nloc] = (__bf16)p1;
            }
        }
        __syncthreads();

        // ---- V fragments (shared), O += P @ V ----
        Frag pf[2];
        #pragma unroll
        for (int t = 0; t < 2; ++t) {
            pf[t].q[0] = *(const v4u*)&Plds[t][nloc][hi * 16];
            pf[t].q[1] = *(const v4u*)&Plds[t][nloc][hi * 16 + 8];
        }
        #pragma unroll
        for (int g = 0; g < 4; ++g) {
            const __bf16* vp = Vt + vbase +
                               (size_t)(g * 16 + nloc) * SEQ + kb + hi * 16;
            Frag vf;
            vf.q[0] = *(const v4u*)(vp);
            vf.q[1] = *(const v4u*)(vp + 8);
            #pragma unroll
            for (int t = 0; t < 2; ++t)
                accO[t][g] = __builtin_amdgcn_wmma_f32_16x16x32_bf16(
                    false, pf[t].v, false, vf.v, (short)0, accO[t][g], false, false);
        }
    }

    // ---- normalize + store (B,N,C) bf16 for the projection GEMM ----
    #pragma unroll
    for (int t = 0; t < 2; ++t) {
        #pragma unroll
        for (int r = 0; r < 8; ++r) {
            float inv = 1.0f / l[t][r];
            int n = q0 + t * 16 + r + 8 * hi;
            size_t orow = (size_t)(b * SEQ + n) * EMB + h * HDIM;
            #pragma unroll
            for (int g = 0; g < 4; ++g)
                AO[orow + g * 16 + nloc] = (__bf16)(accO[t][g][r] * inv);
        }
    }
}

// ---------------------------------------------------------------------------
// out = AO @ Wproj + bproj (fp32 result); one wave -> 16x64 tile (4 accs).
__global__ __launch_bounds__(32)
void proj_kernel(const __bf16* __restrict__ AO,     // ROWS x EMB
                 const __bf16* __restrict__ wT,     // EMB x EMB (transposed)
                 const float*  __restrict__ bias,
                 float* __restrict__ out) {
    const int lane = threadIdx.x;
    const int hi   = lane >> 4;
    const int nloc = lane & 15;
    const int colBase = blockIdx.x * 64;   // [0,768)
    const int rowBase = blockIdx.y * 16;   // [0,4096)

    const __bf16* aptr  = AO + (size_t)(rowBase + nloc) * EMB;
    const __bf16* bptr0 = wT + (size_t)(colBase + nloc) * EMB;
    const int ak = hi * 16, bk = hi * 16;

    v8f acc[4] = {v8f{}, v8f{}, v8f{}, v8f{}};
    for (int kk = 0; kk < EMB; kk += 32) {
        Frag a;
        a.q[0] = *(const v4u*)(aptr + kk + ak);
        a.q[1] = *(const v4u*)(aptr + kk + ak + 8);
        __builtin_prefetch(aptr + kk + 256, 0, 0);
        #pragma unroll
        for (int g = 0; g < 4; ++g) {
            const __bf16* bp = bptr0 + (size_t)g * 16 * EMB + kk + bk;
            Frag b;
            b.q[0] = *(const v4u*)(bp);
            b.q[1] = *(const v4u*)(bp + 8);
            acc[g] = __builtin_amdgcn_wmma_f32_16x16x32_bf16(
                false, a.v, false, b.v, (short)0, acc[g], false, false);
        }
    }

    #pragma unroll
    for (int g = 0; g < 4; ++g) {
        float bc = bias[colBase + g * 16 + nloc];
        #pragma unroll
        for (int r = 0; r < 8; ++r) {
            int row = rowBase + r + 8 * hi;
            out[(size_t)row * EMB + colBase + g * 16 + nloc] = acc[g][r] + bc;
        }
    }
}

// ---------------------------------------------------------------------------
extern "C" void kernel_launch(void* const* d_in, const int* in_sizes, int n_in,
                              void* d_out, int out_size, void* d_ws, size_t ws_size,
                              hipStream_t stream) {
    const float* x     = (const float*)d_in[0];   // 2*2048*768
    const float* cosp  = (const float*)d_in[1];   // 2048*64
    const float* sinp  = (const float*)d_in[2];   // 2048*64
    const float* Wqkv  = (const float*)d_in[3];   // 768*2304
    const float* Wproj = (const float*)d_in[4];   // 768*768
    const float* bproj = (const float*)d_in[5];   // 768

    char* ws = (char*)d_ws;
    // all offsets 16B-aligned
    __bf16* xb     = (__bf16*)(ws + 0);          // 4096x768   bf16  (6,291,456 B)
    __bf16* wqkvT  = (__bf16*)(ws + 6291456);    // 2304x768   bf16  (3,538,944 B)
    __bf16* wprojT = (__bf16*)(ws + 9830400);    // 768x768    bf16  (1,179,648 B)
    __bf16* Qr     = (__bf16*)(ws + 11010048);   // (B,H,N,D)  bf16  (6,291,456 B)
    __bf16* Kr     = (__bf16*)(ws + 17301504);   // (B,H,N,D)  bf16
    __bf16* Vt     = (__bf16*)(ws + 23592960);   // (B,H,D,N)  bf16
    __bf16* AO     = (__bf16*)(ws + 29884416);   // 4096x768   bf16
    // total workspace: 36,175,872 bytes

    const int NX = ROWS * EMB;          // 3,145,728
    f32_to_bf16_kernel<<<(NX + 255) / 256, 256, 0, stream>>>(x, xb, NX);

    const int NW1 = EMB * QKVC;         // 1,769,472
    transpose_to_bf16_kernel<<<(NW1 + 255) / 256, 256, 0, stream>>>(Wqkv, wqkvT, EMB, QKVC);

    const int NW2 = EMB * EMB;          // 589,824
    transpose_to_bf16_kernel<<<(NW2 + 255) / 256, 256, 0, stream>>>(Wproj, wprojT, EMB, EMB);

    qkv_rope_kernel<<<dim3(QKVC / 64, ROWS / 16), 32, 0, stream>>>(
        xb, wqkvT, cosp, sinp, Qr, Kr, Vt);

    attn_kernel<<<dim3(SEQ / 32, BATCH * HEADS), 32, 0, stream>>>(Qr, Kr, Vt, AO);

    proj_kernel<<<dim3(EMB / 64, ROWS / 16), 32, 0, stream>>>(
        AO, wprojT, bproj, (float*)d_out);
}
